// DynamicLayer_88931592831382
// MI455X (gfx1250) — matrix operand order repaired and checked by reading, not compiled, not run on hardware
//
#include <hip/hip_runtime.h>

// ---------------------------------------------------------------------------
// Problem constants (from reference)
// ---------------------------------------------------------------------------
constexpr int BATCH  = 4;
constexpr int SEQ    = 2048;
constexpr int DMODEL = 2048;
constexpr int NH     = 16;
constexpr int NKV    = 4;
constexpr int HDIM   = 128;
constexpr int FFN    = 8192;
constexpr int KSEL   = 1024;          // K = GAMMA * S
constexpr int BKROWS = BATCH * KSEL;  // 4096 selected rows total

// ---------------------------------------------------------------------------
// CDNA5 async VMEM->LDS path (probe via __has_builtin; fallback = plain b128)
// ---------------------------------------------------------------------------
#if defined(__has_builtin)
#if __has_builtin(__builtin_amdgcn_global_load_async_to_lds_b128) && \
    __has_builtin(__builtin_amdgcn_s_wait_asynccnt)
#define HAVE_ASYNC_LDS 1
#endif
#endif
#ifndef HAVE_ASYNC_LDS
#define HAVE_ASYNC_LDS 0
#endif

typedef int v4i __attribute__((ext_vector_type(4)));

// copy 16 bytes global -> LDS (per lane)
__device__ __forceinline__ void cp16_g2l(void* lds, const void* g) {
#if HAVE_ASYNC_LDS
    __builtin_amdgcn_global_load_async_to_lds_b128(
        (__attribute__((address_space(1))) v4i*)g,
        (__attribute__((address_space(3))) v4i*)lds, 0, 0);
#else
    *(uint4*)lds = *(const uint4*)g;
#endif
}

__device__ __forceinline__ void async_stage_fence() {
#if HAVE_ASYNC_LDS
    __builtin_amdgcn_s_wait_asynccnt(0);
#endif
    __syncthreads();
}

// ---------------------------------------------------------------------------
// bf16 helpers / WMMA types
// ---------------------------------------------------------------------------
typedef __bf16 v16bf __attribute__((ext_vector_type(16)));
typedef float  v8f   __attribute__((ext_vector_type(8)));

__device__ __forceinline__ unsigned short f2bfu(float f) {
    unsigned u = __builtin_bit_cast(unsigned, f);
    unsigned r = u + 0x7FFFu + ((u >> 16) & 1u);   // round-to-nearest-even
    return (unsigned short)(r >> 16);
}

__device__ __forceinline__ v8f wmma_bf16(v16bf a, v16bf b, v8f c) {
    return __builtin_amdgcn_wmma_f32_16x16x32_bf16(
        false, a, false, b, (short)0, c, false, false);
}

// A-fragment (16x32 bf16, row-major source, ldk elements per row)
// lane: m = lane&15, half = lane>>4 ; element e -> k = (e<8?0:16)+8*half+(e&7)
__device__ __forceinline__ v16bf load_afrag(const unsigned short* base, int ldk,
                                            int row, int half) {
    v16bf a;
#pragma unroll
    for (int e = 0; e < 16; ++e) {
        int k = ((e < 8) ? 0 : 16) + (half << 3) + (e & 7);
        a[e] = __builtin_bit_cast(__bf16, base[row * ldk + k]);
    }
    return a;
}

// B-fragment (32x16 bf16 from row-major [K][N] source, ldn elements per row)
// lane: n = lane&15, half = lane>>4 ; element e -> k = (e<8?0:16)+8*half+(e&7)
__device__ __forceinline__ v16bf load_bfrag(const unsigned short* base, int ldn,
                                            int col, int half) {
    v16bf b;
#pragma unroll
    for (int e = 0; e < 16; ++e) {
        int k = ((e < 8) ? 0 : 16) + (half << 3) + (e & 7);
        b[e] = __builtin_bit_cast(__bf16, base[k * ldn + col]);
    }
    return b;
}

// ---------------------------------------------------------------------------
// 0. fp32 -> bf16 bulk convert (weights / activations), vectorized x4
// ---------------------------------------------------------------------------
__global__ void f32_to_bf16_kernel(const float4* __restrict__ src,
                                   ushort4* __restrict__ dst, long n4) {
    long i = (long)blockIdx.x * blockDim.x + threadIdx.x;
    long stride = (long)gridDim.x * blockDim.x;
    for (; i < n4; i += stride) {
        float4 v = src[i];
        ushort4 o;
        o.x = f2bfu(v.x);
        o.y = f2bfu(v.y);
        o.z = f2bfu(v.z);
        o.w = f2bfu(v.w);
        dst[i] = o;
    }
}

// ---------------------------------------------------------------------------
// 1. Gate score: g = sigmoid(bce*(mse(post,prior)-ceoff) + bcu*cm*mse(post,orig))
// ---------------------------------------------------------------------------
__global__ __launch_bounds__(256) void gate_kernel(
    const float* __restrict__ post, const float* __restrict__ prior,
    const float* __restrict__ orig, const float* __restrict__ bce,
    const float* __restrict__ bcu, const float* __restrict__ cmult,
    const float* __restrict__ ceoff, float* __restrict__ gOut,
    float* __restrict__ gbOut) {
    __shared__ float r1[256], r2[256];
    int tok = blockIdx.x;
    size_t base = (size_t)tok * DMODEL;
    float se = 0.f, su = 0.f;
    for (int d = threadIdx.x; d < DMODEL; d += 256) {
        float p = post[base + d];
        float a = p - prior[base + d];
        float c = p - orig[base + d];
        se += a * a;
        su += c * c;
    }
    r1[threadIdx.x] = se;
    r2[threadIdx.x] = su;
    __syncthreads();
    for (int o = 128; o > 0; o >>= 1) {
        if (threadIdx.x < o) {
            r1[threadIdx.x] += r1[threadIdx.x + o];
            r2[threadIdx.x] += r2[threadIdx.x + o];
        }
        __syncthreads();
    }
    if (threadIdx.x == 0) {
        float sce = r1[0] / DMODEL, scu = r2[0] / DMODEL;
        float x = bce[0] * (sce - ceoff[0]) + bcu[0] * cmult[0] * scu;
        gOut[tok]  = 1.f / (1.f + __expf(-x));
        gbOut[tok] = 0.f;
    }
}

// ---------------------------------------------------------------------------
// 2. Per-batch top-K via bitonic sort (desc by value, asc index tiebreak),
//    then bitonic index-sort of the K winners. One 1024-thread block / batch.
// ---------------------------------------------------------------------------
__global__ __launch_bounds__(1024) void topk_kernel(const float* __restrict__ g,
                                                    int* __restrict__ idxOut,
                                                    float* __restrict__ gbOut) {
    __shared__ float sv[SEQ];
    __shared__ int   si[SEQ];
    int b = blockIdx.x, tid = threadIdx.x;
    for (int i = tid; i < SEQ; i += 1024) { sv[i] = g[b * SEQ + i]; si[i] = i; }
    __syncthreads();
    for (int k = 2; k <= SEQ; k <<= 1) {
        for (int j = k >> 1; j > 0; j >>= 1) {
            for (int t = tid; t < SEQ; t += 1024) {
                int x = t ^ j;
                if (x > t) {
                    float va = sv[t], vb = sv[x];
                    int ia = si[t], ib = si[x];
                    bool before = (va > vb) || (va == vb && ia < ib);
                    bool up = ((t & k) == 0);
                    if (up ? (!before) : before) {
                        sv[t] = vb; sv[x] = va; si[t] = ib; si[x] = ia;
                    }
                }
            }
            __syncthreads();
        }
    }
    // sort winning indices ascending (first KSEL entries of si)
    for (int k = 2; k <= KSEL; k <<= 1) {
        for (int j = k >> 1; j > 0; j >>= 1) {
            int t = tid, x = t ^ j;
            if (x > t && x < KSEL) {
                int ia = si[t], ib = si[x];
                bool up = ((t & k) == 0);
                if (up ? (ia > ib) : (ia < ib)) { si[t] = ib; si[x] = ia; }
            }
            __syncthreads();
        }
    }
    for (int i = tid; i < KSEL; i += 1024) {
        int v = si[i];
        idxOut[b * KSEL + i] = v;
        gbOut[b * SEQ + v] = 1.0f;
    }
}

// ---------------------------------------------------------------------------
// 3. Copy / gather / scatter
// ---------------------------------------------------------------------------
__global__ void copy_kernel(const float4* __restrict__ src,
                            float4* __restrict__ dst, long n) {
    long i = (long)blockIdx.x * blockDim.x + threadIdx.x;
    long stride = (long)gridDim.x * blockDim.x;
    for (; i < n; i += stride) dst[i] = src[i];
}

__global__ __launch_bounds__(256) void gather_kernel(
    const float* __restrict__ hidden, const int* __restrict__ idx,
    float* __restrict__ sel) {
    int bi = blockIdx.x;                // 0..BKROWS-1
    int b = bi / KSEL;
    int row = idx[bi];
    const float4* src = (const float4*)(hidden + ((size_t)b * SEQ + row) * DMODEL);
    float4* dst = (float4*)(sel + (size_t)bi * DMODEL);
    for (int t = threadIdx.x; t < DMODEL / 4; t += 256) dst[t] = src[t];
}

__global__ __launch_bounds__(256) void scatter_kernel(
    const float* __restrict__ sel, const float* __restrict__ outb,
    const int* __restrict__ idx, const float* __restrict__ g,
    float* __restrict__ finalO) {
    int bi = blockIdx.x;
    int b = bi / KSEL;
    int row = idx[bi];
    float gs = g[b * SEQ + row];
    const float4* s = (const float4*)(sel + (size_t)bi * DMODEL);
    const float4* o = (const float4*)(outb + (size_t)bi * DMODEL);
    float4* f = (float4*)(finalO + ((size_t)b * SEQ + row) * DMODEL);
    for (int t = threadIdx.x; t < DMODEL / 4; t += 256) {
        float4 a = s[t], c = o[t], r;
        r.x = a.x + (c.x - a.x) * gs;
        r.y = a.y + (c.y - a.y) * gs;
        r.z = a.z + (c.z - a.z) * gs;
        r.w = a.w + (c.w - a.w) * gs;
        f[t] = r;
    }
}

// ---------------------------------------------------------------------------
// 4. RMSNorm (one row per block) -> bf16 output (GEMM A operand)
// ---------------------------------------------------------------------------
__global__ __launch_bounds__(256) void rmsnorm_kernel(
    const float* __restrict__ x, const float* __restrict__ w,
    unsigned short* __restrict__ y) {
    __shared__ float red[256];
    int row = blockIdx.x;
    const float* xr = x + (size_t)row * DMODEL;
    float s = 0.f;
    for (int d = threadIdx.x; d < DMODEL; d += 256) { float v = xr[d]; s += v * v; }
    red[threadIdx.x] = s;
    __syncthreads();
    for (int o = 128; o > 0; o >>= 1) {
        if (threadIdx.x < o) red[threadIdx.x] += red[threadIdx.x + o];
        __syncthreads();
    }
    float inv = rsqrtf(red[0] / DMODEL + 1e-6f);
    for (int d = threadIdx.x; d < DMODEL; d += 256)
        y[(size_t)row * DMODEL + d] = f2bfu(xr[d] * inv * w[d]);
}

// ---------------------------------------------------------------------------
// 5. bf16-WMMA GEMM: C[M,N] = A[M,K] * B[K,N] (+ epilogue); A,B bf16 in HBM,
//    staged to LDS with CDNA5 async VMEM->LDS b128 copies.
//    BM=64, BN=128, BK=32. 256 threads = 8 waves in 2x4; wave tile 32x32.
//    DUAL=true: second B matrix, epilogue writes bf16 silu(acc1)*acc2.
// ---------------------------------------------------------------------------
template <bool DUAL>
__global__ __launch_bounds__(256) void gemm_wmma_kernel(
    const unsigned short* __restrict__ A, const unsigned short* __restrict__ B1,
    const unsigned short* __restrict__ B2, float* __restrict__ Cout,
    unsigned short* __restrict__ Tout, const float* __restrict__ bias,
    const float* __restrict__ addm, int M, int N, int Kd, int mode) {
    __shared__ unsigned short As[64][32];
    __shared__ unsigned short Bs[32][128];
    __shared__ unsigned short Bs2[DUAL ? 32 : 1][DUAL ? 128 : 1];

    int tid = threadIdx.x;
    int wid = tid >> 5, lane = tid & 31;
    int wm = wid >> 2, wn = wid & 3;
    int hf = lane >> 4, ln = lane & 15;
    int m0 = blockIdx.y * 64, n0 = blockIdx.x * 128;

    // per-thread staging coordinates (8 bf16 = 16B chunks)
    int ar = tid >> 2, ac = (tid & 3) * 8;           // A: 64x32 -> 256 chunks
    int br0 = tid >> 4, bc0 = (tid & 15) * 8;        // B: 32x128 -> 512 chunks
    int br1 = (tid + 256) >> 4, bc1 = ((tid + 256) & 15) * 8;

    v8f acc[2][2];
    v8f acc2[2][2];
#pragma unroll
    for (int i = 0; i < 2; ++i)
#pragma unroll
        for (int j = 0; j < 2; ++j)
#pragma unroll
            for (int r = 0; r < 8; ++r) {
                acc[i][j][r] = 0.f;
                if (DUAL) acc2[i][j][r] = 0.f;
            }

    for (int k0 = 0; k0 < Kd; k0 += 32) {
        cp16_g2l(&As[ar][ac], A + (size_t)(m0 + ar) * Kd + k0 + ac);
        cp16_g2l(&Bs[br0][bc0], B1 + (size_t)(k0 + br0) * N + n0 + bc0);
        cp16_g2l(&Bs[br1][bc1], B1 + (size_t)(k0 + br1) * N + n0 + bc1);
        if (DUAL) {
            cp16_g2l(&Bs2[br0][bc0], B2 + (size_t)(k0 + br0) * N + n0 + bc0);
            cp16_g2l(&Bs2[br1][bc1], B2 + (size_t)(k0 + br1) * N + n0 + bc1);
        }
        async_stage_fence();

        v16bf a0 = load_afrag(&As[0][0], 32, wm * 32 + ln, hf);
        v16bf a1 = load_afrag(&As[0][0], 32, wm * 32 + 16 + ln, hf);
        v16bf b0 = load_bfrag(&Bs[0][0], 128, wn * 32 + ln, hf);
        v16bf b1 = load_bfrag(&Bs[0][0], 128, wn * 32 + 16 + ln, hf);
        acc[0][0] = wmma_bf16(a0, b0, acc[0][0]);
        acc[0][1] = wmma_bf16(a0, b1, acc[0][1]);
        acc[1][0] = wmma_bf16(a1, b0, acc[1][0]);
        acc[1][1] = wmma_bf16(a1, b1, acc[1][1]);
        if (DUAL) {
            v16bf c0 = load_bfrag(&Bs2[0][0], 128, wn * 32 + ln, hf);
            v16bf c1 = load_bfrag(&Bs2[0][0], 128, wn * 32 + 16 + ln, hf);
            acc2[0][0] = wmma_bf16(a0, c0, acc2[0][0]);
            acc2[0][1] = wmma_bf16(a0, c1, acc2[0][1]);
            acc2[1][0] = wmma_bf16(a1, c0, acc2[1][0]);
            acc2[1][1] = wmma_bf16(a1, c1, acc2[1][1]);
        }
        __syncthreads();
    }

#pragma unroll
    for (int i = 0; i < 2; ++i)
#pragma unroll
        for (int j = 0; j < 2; ++j) {
            int col = n0 + wn * 32 + j * 16 + ln;
#pragma unroll
            for (int r = 0; r < 8; ++r) {
                int row = m0 + wm * 32 + i * 16 + r + 8 * hf;
                float v = acc[i][j][r];
                if (DUAL) {
                    float u = acc2[i][j][r];
                    float t = (v / (1.f + __expf(-v))) * u;  // silu(g)*u
                    Tout[(size_t)row * N + col] = f2bfu(t);
                } else if (mode == 0) {
                    Cout[(size_t)row * N + col] = v + (bias ? bias[col] : 0.f);
                } else {
                    Cout[(size_t)row * N + col] = v + addm[(size_t)row * N + col];
                }
            }
        }
}

// ---------------------------------------------------------------------------
// 6. RoPE + bf16 pack:  src [BKROWS][nheads*HDIM] f32 ->
//    dst [BATCH][nheads][KSEL][HDIM] bf16. pos = selected index value.
// ---------------------------------------------------------------------------
__global__ void rope_pack_kernel(const float* __restrict__ src,
                                 const int* __restrict__ idx,
                                 unsigned short* __restrict__ dst, int nheads,
                                 int do_rope) {
    int gid = blockIdx.x * blockDim.x + threadIdx.x;
    int total = BKROWS * nheads * (HDIM / 2);
    if (gid >= total) return;
    int j = gid & 63;
    int h = (gid >> 6) % nheads;
    int bi = gid / (64 * nheads);
    int b = bi / KSEL, i = bi % KSEL;
    const float* x = src + (size_t)bi * nheads * HDIM + h * HDIM;
    float x1 = x[j], x2 = x[j + 64];
    float o1, o2;
    if (do_rope) {
        float pos = (float)idx[bi];
        float inv = __expf(-(2.f * j / 128.f) * 9.210340371976184f); // ln(1e4)
        float ang = pos * inv;
        float c = __cosf(ang), s = __sinf(ang);
        o1 = x1 * c - x2 * s;
        o2 = x2 * c + x1 * s;
    } else {
        o1 = x1; o2 = x2;
    }
    unsigned short* d = dst + (((size_t)(b * nheads + h) * KSEL) + i) * HDIM;
    d[j] = f2bfu(o1);
    d[j + 64] = f2bfu(o2);
}

// ---------------------------------------------------------------------------
// 7. Flash attention (GQA, causal over selected positions).
//    Block = 256 threads = 8 waves; each wave owns 16 query rows;
//    block covers 128 q rows of one (b,h). Q.K^T and P.V on WMMA.
//    K/V tiles staged with async VMEM->LDS. ctx written as bf16 (GEMM A).
// ---------------------------------------------------------------------------
__global__ __launch_bounds__(256) void attention_kernel(
    const unsigned short* __restrict__ qb, const unsigned short* __restrict__ kb,
    const unsigned short* __restrict__ vb, unsigned short* __restrict__ ctxb) {
    __shared__ unsigned short Ks[32][HDIM];
    __shared__ unsigned short Vs[32][HDIM];
    __shared__ unsigned short Ps[8][16][32];

    int wg = blockIdx.x;
    int qblk = wg & 7;
    int h = (wg >> 3) & 15;
    int b = wg >> 7;
    int kvh = h >> 2;
    int tid = threadIdx.x, wid = tid >> 5, lane = tid & 31;
    int hf = lane >> 4, ln = lane & 15;
    int q0 = qblk * 128;
    int qrow_base = q0 + wid * 16;

    const unsigned short* Qb = qb + ((size_t)(b * NH + h)) * KSEL * HDIM;
    const unsigned short* Kb = kb + ((size_t)(b * NKV + kvh)) * KSEL * HDIM;
    const unsigned short* Vb = vb + ((size_t)(b * NKV + kvh)) * KSEL * HDIM;

    v16bf aQ[4];
#pragma unroll
    for (int st = 0; st < 4; ++st) {
        v16bf a;
#pragma unroll
        for (int e = 0; e < 16; ++e) {
            int k = ((e < 8) ? 0 : 16) + (hf << 3) + (e & 7);
            a[e] = __builtin_bit_cast(
                __bf16, Qb[(size_t)(qrow_base + ln) * HDIM + st * 32 + k]);
        }
        aQ[st] = a;
    }

    float mi[8], li[8];
    v8f accO[8];
#pragma unroll
    for (int r = 0; r < 8; ++r) { mi[r] = -1e30f; li[r] = 0.f; }
#pragma unroll
    for (int t = 0; t < 8; ++t)
#pragma unroll
        for (int r = 0; r < 8; ++r) accO[t][r] = 0.f;

    const float scl = 0.08838834764831845f;  // 1/sqrt(128)
    int nblk = (q0 + 128) >> 5;

    for (int jb = 0; jb < nblk; ++jb) {
        __syncthreads();
        // stage 32 keys + 32 values (each 32x128 bf16 = 8KB, contiguous rows)
        {
            unsigned short* ksf = &Ks[0][0];
            unsigned short* vsf = &Vs[0][0];
            int l0 = tid, l1 = tid + 256;   // 512 chunks of 8 bf16 per tile
            cp16_g2l(ksf + l0 * 8, Kb + (size_t)jb * 4096 + l0 * 8);
            cp16_g2l(ksf + l1 * 8, Kb + (size_t)jb * 4096 + l1 * 8);
            cp16_g2l(vsf + l0 * 8, Vb + (size_t)jb * 4096 + l0 * 8);
            cp16_g2l(vsf + l1 * 8, Vb + (size_t)jb * 4096 + l1 * 8);
        }
        async_stage_fence();

        v8f s0, s1;
#pragma unroll
        for (int r = 0; r < 8; ++r) { s0[r] = 0.f; s1[r] = 0.f; }
#pragma unroll
        for (int st = 0; st < 4; ++st) {
            v16bf b0, b1;
#pragma unroll
            for (int e = 0; e < 16; ++e) {
                int kk = st * 32 + ((e < 8) ? 0 : 16) + (hf << 3) + (e & 7);
                b0[e] = __builtin_bit_cast(__bf16, Ks[ln][kk]);
                b1[e] = __builtin_bit_cast(__bf16, Ks[16 + ln][kk]);
            }
            s0 = wmma_bf16(aQ[st], b0, s0);
            s1 = wmma_bf16(aQ[st], b1, s1);
        }

        float p0[8], p1[8], sc[8];
#pragma unroll
        for (int r = 0; r < 8; ++r) {
            int qrow = qrow_base + r + 8 * hf;
            int k0i = jb * 32 + ln;
            int k1i = jb * 32 + 16 + ln;
            float v0 = (k0i <= qrow) ? s0[r] * scl : -1e30f;
            float v1 = (k1i <= qrow) ? s1[r] * scl : -1e30f;
            float mx = fmaxf(v0, v1);
#pragma unroll
            for (int off = 8; off >= 1; off >>= 1)
                mx = fmaxf(mx, __shfl_xor(mx, off));
            float mnew = fmaxf(mi[r], mx);
            float s = __expf(mi[r] - mnew);
            float e0 = __expf(v0 - mnew), e1 = __expf(v1 - mnew);
            float rs = e0 + e1;
#pragma unroll
            for (int off = 8; off >= 1; off >>= 1) rs += __shfl_xor(rs, off);
            li[r] = li[r] * s + rs;
            mi[r] = mnew;
            sc[r] = s;
            p0[r] = e0;
            p1[r] = e1;
        }
#pragma unroll
        for (int t = 0; t < 8; ++t)
#pragma unroll
            for (int r = 0; r < 8; ++r) accO[t][r] *= sc[r];

#pragma unroll
        for (int r = 0; r < 8; ++r) {
            Ps[wid][r + 8 * hf][ln] = f2bfu(p0[r]);
            Ps[wid][r + 8 * hf][16 + ln] = f2bfu(p1[r]);
        }
        __syncthreads();

        v16bf aP;
#pragma unroll
        for (int e = 0; e < 16; ++e) {
            int k = ((e < 8) ? 0 : 16) + (hf << 3) + (e & 7);
            aP[e] = __builtin_bit_cast(__bf16, Ps[wid][ln][k]);
        }
#pragma unroll
        for (int t = 0; t < 8; ++t) {
            v16bf bV;
#pragma unroll
            for (int e = 0; e < 16; ++e) {
                int kk = ((e < 8) ? 0 : 16) + (hf << 3) + (e & 7);
                bV[e] = __builtin_bit_cast(__bf16, Vs[kk][t * 16 + ln]);
            }
            accO[t] = wmma_bf16(aP, bV, accO[t]);
        }
    }

#pragma unroll
    for (int t = 0; t < 8; ++t)
#pragma unroll
        for (int r = 0; r < 8; ++r) {
            int grow = b * KSEL + qrow_base + r + 8 * hf;
            ctxb[(size_t)grow * (NH * HDIM) + h * HDIM + t * 16 + ln] =
                f2bfu(accO[t][r] / li[r]);
        }
}

// ---------------------------------------------------------------------------
// Host-side launcher
// ---------------------------------------------------------------------------
extern "C" void kernel_launch(void* const* d_in, const int* in_sizes, int n_in,
                              void* d_out, int out_size, void* d_ws,
                              size_t ws_size, hipStream_t stream) {
    const float* hidden = (const float*)d_in[0];
    const float* vorig  = (const float*)d_in[1];
    const float* vpost  = (const float*)d_in[2];
    const float* vprior = (const float*)d_in[3];
    // d_in[4] position_ids == arange(S): positions equal the gathered indices.
    const float* ln1 = (const float*)d_in[5];
    const float* ln2 = (const float*)d_in[6];
    const float* wq = (const float*)d_in[7];
    const float* bq = (const float*)d_in[8];
    const float* wk = (const float*)d_in[9];
    const float* bk = (const float*)d_in[10];
    const float* wv = (const float*)d_in[11];
    const float* bv = (const float*)d_in[12];
    const float* wo = (const float*)d_in[13];
    const float* wgate = (const float*)d_in[14];
    const float* wup = (const float*)d_in[15];
    const float* wdown = (const float*)d_in[16];
    const float* bce = (const float*)d_in[17];
    const float* bcu = (const float*)d_in[18];
    const float* cmult = (const float*)d_in[19];
    const float* ceoff = (const float*)d_in[20];

    float* out_final = (float*)d_out;
    float* out_g = out_final + (size_t)BATCH * SEQ * DMODEL;
    float* out_gb = out_g + (size_t)BATCH * SEQ;

    char* p = (char*)d_ws;
    auto carve = [&](size_t bytes) -> void* {
        void* r = (void*)p;
        p += (bytes + 255) & ~(size_t)255;
        return r;
    };
    int* idxW = (int*)carve((size_t)BATCH * KSEL * sizeof(int));
    float* sel = (float*)carve((size_t)BKROWS * DMODEL * 4);
    unsigned short* bufA = (unsigned short*)carve((size_t)BKROWS * DMODEL * 2);
    float* qbuf = (float*)carve((size_t)BKROWS * NH * HDIM * 4);  // also outb
    float* kfbuf = (float*)carve((size_t)BKROWS * NKV * HDIM * 4);
    float* vfbuf = (float*)carve((size_t)BKROWS * NKV * HDIM * 4);
    unsigned short* qbb = (unsigned short*)carve((size_t)BKROWS * NH * HDIM * 2);
    unsigned short* kbb = (unsigned short*)carve((size_t)BKROWS * NKV * HDIM * 2);
    unsigned short* vbb = (unsigned short*)carve((size_t)BKROWS * NKV * HDIM * 2);
    unsigned short* ctxb = (unsigned short*)carve((size_t)BKROWS * NH * HDIM * 2);
    float* sel2 = (float*)carve((size_t)BKROWS * DMODEL * 4);
    unsigned short* tbuf = (unsigned short*)carve((size_t)BKROWS * FFN * 2);
    // bf16 weight copies
    unsigned short* wqb = (unsigned short*)carve((size_t)DMODEL * DMODEL * 2);
    unsigned short* wkb = (unsigned short*)carve((size_t)DMODEL * NKV * HDIM * 2);
    unsigned short* wvb = (unsigned short*)carve((size_t)DMODEL * NKV * HDIM * 2);
    unsigned short* wob = (unsigned short*)carve((size_t)DMODEL * DMODEL * 2);
    unsigned short* wgb = (unsigned short*)carve((size_t)DMODEL * FFN * 2);
    unsigned short* wub = (unsigned short*)carve((size_t)DMODEL * FFN * 2);
    unsigned short* wdb = (unsigned short*)carve((size_t)FFN * DMODEL * 2);
    float* outb = qbuf;  // reuse q fp32 buffer for down-proj output

    // 0. weight conversions fp32 -> bf16 (one streaming pass, ~15us at HBM BW)
    auto conv = [&](const float* s, unsigned short* d, long n) {
        long n4 = n / 4;
        int blocks = (int)((n4 + 255) / 256);
        if (blocks > 4096) blocks = 4096;
        f32_to_bf16_kernel<<<blocks, 256, 0, stream>>>((const float4*)s,
                                                       (ushort4*)d, n4);
    };
    conv(wq, wqb, (long)DMODEL * DMODEL);
    conv(wk, wkb, (long)DMODEL * NKV * HDIM);
    conv(wv, wvb, (long)DMODEL * NKV * HDIM);
    conv(wo, wob, (long)DMODEL * DMODEL);
    conv(wgate, wgb, (long)DMODEL * FFN);
    conv(wup, wub, (long)DMODEL * FFN);
    conv(wdown, wdb, (long)FFN * DMODEL);

    // 1. gating + zero gate_binary
    gate_kernel<<<BATCH * SEQ, 256, 0, stream>>>(vpost, vprior, vorig, bce, bcu,
                                                 cmult, ceoff, out_g, out_gb);
    // 2. top-K + idx sort + gate_binary ones
    topk_kernel<<<BATCH, 1024, 0, stream>>>(out_g, idxW, out_gb);
    // 3. final = hidden (selected rows overwritten later); gather sel
    copy_kernel<<<8192, 256, 0, stream>>>((const float4*)hidden,
                                          (float4*)out_final,
                                          (long)BATCH * SEQ * DMODEL / 4);
    gather_kernel<<<BKROWS, 256, 0, stream>>>(hidden, idxW, sel);
    // 4. RMSNorm 1 -> bf16
    rmsnorm_kernel<<<BKROWS, 256, 0, stream>>>(sel, ln1, bufA);
    // 5. QKV projections (bias add epilogue)
    gemm_wmma_kernel<false><<<dim3(DMODEL / 128, BKROWS / 64), 256, 0, stream>>>(
        bufA, wqb, nullptr, qbuf, nullptr, bq, nullptr, BKROWS, DMODEL, DMODEL,
        0);
    gemm_wmma_kernel<false><<<dim3((NKV * HDIM) / 128, BKROWS / 64), 256, 0,
                              stream>>>(bufA, wkb, nullptr, kfbuf, nullptr, bk,
                                        nullptr, BKROWS, NKV * HDIM, DMODEL, 0);
    gemm_wmma_kernel<false><<<dim3((NKV * HDIM) / 128, BKROWS / 64), 256, 0,
                              stream>>>(bufA, wvb, nullptr, vfbuf, nullptr, bv,
                                        nullptr, BKROWS, NKV * HDIM, DMODEL, 0);
    // 6. RoPE + pack to [B, head, seq, HD] bf16
    {
        int tq = BKROWS * NH * 64, tk = BKROWS * NKV * 64;
        rope_pack_kernel<<<(tq + 255) / 256, 256, 0, stream>>>(qbuf, idxW, qbb,
                                                               NH, 1);
        rope_pack_kernel<<<(tk + 255) / 256, 256, 0, stream>>>(kfbuf, idxW, kbb,
                                                               NKV, 1);
        rope_pack_kernel<<<(tk + 255) / 256, 256, 0, stream>>>(vfbuf, idxW, vbb,
                                                               NKV, 0);
    }
    // 7. flash attention -> ctx bf16 [BKROWS][NH*HDIM]
    attention_kernel<<<BATCH * NH * (KSEL / 128), 256, 0, stream>>>(qbb, kbb,
                                                                    vbb, ctxb);
    // 8. O-projection + residual: sel2 = sel + ctx @ wo
    gemm_wmma_kernel<false><<<dim3(DMODEL / 128, BKROWS / 64), 256, 0, stream>>>(
        ctxb, wob, nullptr, sel2, nullptr, nullptr, sel, BKROWS, DMODEL, DMODEL,
        1);
    // 9. RMSNorm 2 -> bf16
    rmsnorm_kernel<<<BKROWS, 256, 0, stream>>>(sel2, ln2, bufA);
    // 10. fused SwiGLU: t = silu(h2@Wg) * (h2@Wu) -> bf16
    gemm_wmma_kernel<true><<<dim3(FFN / 128, BKROWS / 64), 256, 0, stream>>>(
        bufA, wgb, wub, nullptr, tbuf, nullptr, nullptr, BKROWS, FFN, DMODEL, 2);
    // 11. down-projection + residual: outb = sel2 + t @ Wd
    gemm_wmma_kernel<false><<<dim3(DMODEL / 128, BKROWS / 64), 256, 0, stream>>>(
        tbuf, wdb, nullptr, outb, nullptr, nullptr, sel2, BKROWS, DMODEL, FFN,
        1);
    // 12. gated scatter into final
    scatter_kernel<<<BKROWS, 256, 0, stream>>>(sel, outb, idxW, out_g,
                                               out_final);
}
